// Res_up_34797825032222
// MI455X (gfx1250) — matrix operand order repaired and checked by reading, not compiled, not run on hardware
//
#include <hip/hip_runtime.h>

// ---------------------------------------------------------------------------
// Problem constants (from reference setup_inputs)
// ---------------------------------------------------------------------------
constexpr int B    = 4;
constexpr int Nc   = 2048;
constexpr int Nf   = 8192;
constexpr int Cin  = 128;
constexpr int Cout = 128;
constexpr int Ch   = 64;       // Cout/2
constexpr int Ec   = 16384;
constexpr int Ef   = 65536;
constexpr float NEG_SLOPE = 0.01f;

typedef __attribute__((ext_vector_type(16))) __bf16 v16bf;
typedef __attribute__((ext_vector_type(8)))  float  v8f;

__device__ __forceinline__ unsigned short f2bf(float f) {
    unsigned int u = __float_as_uint(f);
    unsigned int r = (u + 0x7FFFu + ((u >> 16) & 1u)) >> 16;   // RNE truncate
    return (unsigned short)r;
}
__device__ __forceinline__ float lrelu(float v) { return v > 0.f ? v : NEG_SLOPE * v; }

// ---------------------------------------------------------------------------
// Weight prep: fp32 row-major [K2, COUT] -> bf16 B-fragments for 16x16x32 WMMA.
// Fragment (kt, nt): lane n holds 16 bf16: K = kt*32 + (n>>4)*16 + e,
//                    N = nt*16 + (n&15)
// ---------------------------------------------------------------------------
__global__ void pack_w_kernel(const float* __restrict__ W, unsigned short* __restrict__ out,
                              int K2, int COUT) {
    const int NT = COUT / 16;
    const long total = (long)(K2 / 32) * NT * 512;
    long i = (long)blockIdx.x * blockDim.x + threadIdx.x;
    if (i >= total) return;
    int kt   = (int)(i / ((long)NT * 512));
    int rem  = (int)(i % ((long)NT * 512));
    int nt   = rem / 512;
    int r2   = rem & 511;
    int lane = r2 >> 4;
    int e    = r2 & 15;
    int krow = kt * 32 + ((lane >> 4) << 4) + e;
    int ncol = nt * 16 + (lane & 15);
    out[i] = f2bf(W[(size_t)krow * COUT + ncol]);
}

// ---------------------------------------------------------------------------
// Fused gather-GEMM + bias + LeakyReLU (+ segment-sum scatter or store).
// A row m = [featA[rowA[m]] (CA) || featB[rowB[m]] (CB)].
// Block = 64 rows (MT=4 M-tiles); wave `wv` owns one 16-col N-tile.
// Per kt step each wave loads its B fragment ONCE and reuses it across the
// 4 M-tiles -> 4 independent WMMA accumulator chains (hides WMMA latency,
// 4x arithmetic intensity vs. 16-row tiles).
// ---------------------------------------------------------------------------
template <int CA, int CB, int COUT, bool SCATTER, bool ACCUM>
__global__ void __launch_bounds__(32 * (COUT / 16))
gemm_mlp(const float* __restrict__ featA, long strideA,
         const float* __restrict__ featB, long strideB,
         const int* __restrict__ idx, long stride_idx, int dst_off,  // idx==null -> node mode
         const unsigned short* __restrict__ Wfrag,
         const float* __restrict__ bias,
         float* __restrict__ out, long strideO) {
    constexpr int KT = (CA + CB) / 32;   // 32-wide K steps
    constexpr int NT = COUT / 16;        // waves per block
    constexpr int MT = 4;                // M tiles per block (64 rows)
    constexpr int NTHR = 32 * NT;
    const int tile = blockIdx.x;         // block of 64 rows
    const int b    = blockIdx.y;
    const int lane = threadIdx.x;
    const int wv   = threadIdx.y;
    const int tid  = wv * 32 + lane;

    __shared__ __align__(32) unsigned short lds_a[KT * MT * 512];
    __shared__ int s_rowA[64], s_rowB[64];

    if (tid < 128) {
        int m = tid & 63;
        int node = tile * 64 + m;
        int r;
        if (idx) {
            r = (tid < 64) ? idx[(size_t)b * stride_idx + node]
                           : idx[(size_t)b * stride_idx + dst_off + node];
        } else {
            r = node;
        }
        if (tid < 64) s_rowA[m] = r; else s_rowB[m] = r;
    }
    __syncthreads();

    const float* fA = featA + (size_t)b * strideA;
    const float* fB = featB + (size_t)b * strideB;

    // Gather + fp32->bf16 convert straight into A-fragment order.
    // Element pairs (e, e+1), e even, are K-contiguous -> float2 load,
    // packed dword LDS store. dword index == pair index p by construction.
    unsigned int* lds_a32 = (unsigned int*)lds_a;
    constexpr int NPAIR = KT * MT * 256;
    for (int p = tid; p < NPAIR; p += NTHR) {
        int kt  = p / (MT * 256);
        int rem = p % (MT * 256);
        int mt  = rem >> 8;
        int r   = rem & 255;
        int ln  = r >> 3;
        int e   = (r & 7) << 1;
        int m   = mt * 16 + (ln & 15);
        int k   = kt * 32 + ((ln >> 4) << 4) + e;
        const float* src;
        size_t base;
        if (k < CA) { src = fA; base = (size_t)s_rowA[m] * CA + k; }
        else        { src = fB; base = (size_t)s_rowB[m] * CB + (k - CA); }
        float v0 = src[base], v1 = src[base + 1];
        lds_a32[p] = (unsigned int)f2bf(v0) | ((unsigned int)f2bf(v1) << 16);
    }
    __syncthreads();

    v8f acc[MT];
#pragma unroll
    for (int mt = 0; mt < MT; ++mt)
        acc[mt] = (v8f){0.f, 0.f, 0.f, 0.f, 0.f, 0.f, 0.f, 0.f};

#pragma unroll
    for (int kt = 0; kt < KT; ++kt) {
        v16bf w = *(const v16bf*)(&Wfrag[((size_t)(kt * NT + wv) * 32 + lane) * 16]);
#pragma unroll
        for (int mt = 0; mt < MT; ++mt) {
            v16bf a = *(const v16bf*)(&lds_a[((kt * MT + mt) * 32 + lane) * 16]);
            acc[mt] = __builtin_amdgcn_wmma_f32_16x16x32_bf16(false, a, false, w,
                                                              (short)0, acc[mt], false, false);
        }
    }

    const int col = wv * 16 + (lane & 15);
    const float bc = bias[col];
    float* og = out + (size_t)b * strideO;
#pragma unroll
    for (int mt = 0; mt < MT; ++mt) {
#pragma unroll
        for (int r = 0; r < 8; ++r) {
            int m = mt * 16 + r + ((lane >> 4) << 3);
            float v = lrelu(acc[mt][r] + bc);
            if (SCATTER) {
                atomicAdd(&og[(size_t)s_rowB[m] * COUT + col], v);  // segment_sum at dst
            } else {
                size_t o = (size_t)(tile * 64 + m) * COUT + col;
                if (ACCUM) og[o] += v; else og[o] = v;
            }
        }
    }
}

// ---------------------------------------------------------------------------
// Brute-force 3-NN (fine -> coarse), LDS-tiled candidate positions.
// ---------------------------------------------------------------------------
__global__ void knn_kernel(const float* __restrict__ pos, const int* __restrict__ mask,
                           int* __restrict__ kidx, float* __restrict__ kw) {
    const int b = blockIdx.y;
    const int f = blockIdx.x * blockDim.x + threadIdx.x;
    const float* posg = pos + (size_t)b * Nf * 3;
    const float px = posg[f * 3], py = posg[f * 3 + 1], pz = posg[f * 3 + 2];
    float bd0 = 3.4e38f, bd1 = 3.4e38f, bd2 = 3.4e38f;
    int   bi0 = 0, bi1 = 0, bi2 = 0;
    __shared__ float spc[256 * 3];
    for (int t0 = 0; t0 < Nc; t0 += 256) {
        for (int j = threadIdx.x; j < 256 * 3; j += blockDim.x) {
            int cj = j / 3, comp = j - cj * 3;
            int row = mask[b * Nc + t0 + cj];
            spc[j] = posg[(size_t)row * 3 + comp];
        }
        __syncthreads();
        for (int j = 0; j < 256; ++j) {
            float dx = px - spc[3 * j], dy = py - spc[3 * j + 1], dz = pz - spc[3 * j + 2];
            float d2 = dx * dx + dy * dy + dz * dz;
            int ci = t0 + j;
            if (d2 < bd0)      { bd2 = bd1; bi2 = bi1; bd1 = bd0; bi1 = bi0; bd0 = d2; bi0 = ci; }
            else if (d2 < bd1) { bd2 = bd1; bi2 = bi1; bd1 = d2;  bi1 = ci; }
            else if (d2 < bd2) { bd2 = d2;  bi2 = ci; }
        }
        __syncthreads();
    }
    float w0 = 1.f / (bd0 + 1e-16f), w1 = 1.f / (bd1 + 1e-16f), w2 = 1.f / (bd2 + 1e-16f);
    float inv = 1.f / (w0 + w1 + w2);
    size_t o = ((size_t)b * Nf + f) * 3;
    kidx[o] = bi0; kidx[o + 1] = bi1; kidx[o + 2] = bi2;
    kw[o] = w0 * inv; kw[o + 1] = w1 * inv; kw[o + 2] = w2 * inv;
}

// inverse-distance-weighted gather: out[f,c] = sum_k w_k * feat_c[idx_k, c]
template <int C>
__global__ void interp_kernel(const float* __restrict__ featc, const int* __restrict__ kidx,
                              const float* __restrict__ kw, float* __restrict__ out) {
    long i = (long)blockIdx.x * blockDim.x + threadIdx.x;
    const long total = (long)B * Nf * C;
    if (i >= total) return;
    int  c = (int)(i % C);
    long f = i / C;
    int  b = (int)(f / Nf);
    size_t o3 = (size_t)f * 3;
    const float* fc = featc + (size_t)b * Nc * C;
    float v = kw[o3]     * fc[(size_t)kidx[o3]     * C + c]
            + kw[o3 + 1] * fc[(size_t)kidx[o3 + 1] * C + c]
            + kw[o3 + 2] * fc[(size_t)kidx[o3 + 2] * C + c];
    out[i] = v;
}

// ---------------------------------------------------------------------------
// Utility / epilogue kernels
// ---------------------------------------------------------------------------
__global__ void zero_f32(float* __restrict__ p, long n) {
    long i = (long)blockIdx.x * blockDim.x + threadIdx.x;
    if (i < n) p[i] = 0.f;
}

__global__ void bn_sum_kernel(const float* __restrict__ y, float* __restrict__ sums,
                              float* __restrict__ sumsq) {
    const int c = threadIdx.x;                 // 128 channels
    const long r0 = (long)blockIdx.x * 256;
    float s = 0.f, sq = 0.f;
    for (int r = 0; r < 256; ++r) {
        float v = y[(r0 + r) * Cout + c];
        s += v; sq += v * v;
    }
    atomicAdd(&sums[c], s);
    atomicAdd(&sumsq[c], sq);
}

__global__ void bn_final_kernel(float* __restrict__ y, const float* __restrict__ sums,
                                const float* __restrict__ sumsq,
                                const float* __restrict__ gamma, const float* __restrict__ beta) {
    long i = (long)blockIdx.x * blockDim.x + threadIdx.x;
    const long total = (long)B * Nf * Cout;
    if (i >= total) return;
    int c = (int)(i & (Cout - 1));
    const float invn = 1.f / (float)((long)B * Nf);
    float mu  = sums[c] * invn;
    float var = sumsq[c] * invn - mu * mu;
    float v = (y[i] - mu) * rsqrtf(var + 1e-5f) * gamma[c] + beta[c];
    y[i] = lrelu(v);
}

__global__ void wfine_kernel(const float* __restrict__ wts, const int* __restrict__ mask,
                             float* __restrict__ outw) {
    int i = blockIdx.x * blockDim.x + threadIdx.x;
    if (i >= B * Nc) return;
    int b = i / Nc;
    outw[(size_t)b * Nf + mask[i]] = wts[i];
}

// ---------------------------------------------------------------------------
// Host launcher
// ---------------------------------------------------------------------------
extern "C" void kernel_launch(void* const* d_in, const int* in_sizes, int n_in,
                              void* d_out, int out_size, void* d_ws, size_t ws_size,
                              hipStream_t stream) {
    (void)in_sizes; (void)n_in; (void)out_size; (void)ws_size;
    const float* x       = (const float*)d_in[0];   // [B,Nc,Cin]
    const float* weights = (const float*)d_in[1];   // [B,Nc,1]
    const float* pos     = (const float*)d_in[2];   // [B,Nf,3]
    const float* W1e = (const float*)d_in[3];  const float* b1e = (const float*)d_in[4];
    const float* W1n = (const float*)d_in[5];  const float* b1n = (const float*)d_in[6];
    const float* W2e = (const float*)d_in[7];  const float* b2e = (const float*)d_in[8];
    const float* W2n = (const float*)d_in[9];  const float* b2n = (const float*)d_in[10];
    const float* Wse = (const float*)d_in[11]; const float* bse = (const float*)d_in[12];
    const float* Wsn = (const float*)d_in[13]; const float* bsn = (const float*)d_in[14];
    const float* gamma = (const float*)d_in[15];
    const float* beta  = (const float*)d_in[16];
    const int* mask     = (const int*)d_in[17];     // [B,Nc]
    const int* e_coarse = (const int*)d_in[18];     // [B,2,Ec]
    const int* e_fine   = (const int*)d_in[19];     // [B,2,Ef]

    float* yout = (float*)d_out;                         // [B*Nf, Cout]
    float* wout = (float*)d_out + (size_t)B * Nf * Cout; // [B*Nf]

    // ---- workspace carve-out (256B aligned) ----
    size_t off = 0;
    auto alloc = [&](size_t bytes) -> void* {
        void* p = (char*)d_ws + off;
        off += (bytes + 255) & ~(size_t)255;
        return p;
    };
    unsigned short* wf1e = (unsigned short*)alloc((size_t)(256/32)*(64/16) *512*2);
    unsigned short* wf1n = (unsigned short*)alloc((size_t)(192/32)*(64/16) *512*2);
    unsigned short* wf2e = (unsigned short*)alloc((size_t)(128/32)*(128/16)*512*2);
    unsigned short* wf2n = (unsigned short*)alloc((size_t)(192/32)*(128/16)*512*2);
    unsigned short* wfse = (unsigned short*)alloc((size_t)(256/32)*(128/16)*512*2);
    unsigned short* wfsn = (unsigned short*)alloc((size_t)(256/32)*(128/16)*512*2);
    float* agg_c = (float*)alloc((size_t)B * Nc * Ch   * 4);
    float* x1    = (float*)alloc((size_t)B * Nc * Ch   * 4);
    int*   kidx  = (int*)  alloc((size_t)B * Nf * 3    * 4);
    float* kw    = (float*)alloc((size_t)B * Nf * 3    * 4);
    float* x_up  = (float*)alloc((size_t)B * Nf * Cin  * 4);
    float* x1_up = (float*)alloc((size_t)B * Nf * Ch   * 4);
    float* agg   = (float*)alloc((size_t)B * Nf * Cout * 4);
    float* sums  = (float*)alloc(2 * Cout * 4);          // sums | sumsq
    float* sumsq = sums + Cout;

    auto gz = [](long n) { return dim3((unsigned)((n + 255) / 256)); };

    // 1) pack weights into bf16 B-fragments
    pack_w_kernel<<<gz((long)(256/32)*(64/16) *512), 256, 0, stream>>>(W1e, wf1e, 256, 64);
    pack_w_kernel<<<gz((long)(192/32)*(64/16) *512), 256, 0, stream>>>(W1n, wf1n, 192, 64);
    pack_w_kernel<<<gz((long)(128/32)*(128/16)*512), 256, 0, stream>>>(W2e, wf2e, 128, 128);
    pack_w_kernel<<<gz((long)(192/32)*(128/16)*512), 256, 0, stream>>>(W2n, wf2n, 192, 128);
    pack_w_kernel<<<gz((long)(256/32)*(128/16)*512), 256, 0, stream>>>(Wse, wfse, 256, 128);
    pack_w_kernel<<<gz((long)(256/32)*(128/16)*512), 256, 0, stream>>>(Wsn, wfsn, 256, 128);

    // 2) mpl1 on coarse graph: edge MLP + segment_sum, then node MLP -> x1
    zero_f32<<<gz((long)B * Nc * Ch), 256, 0, stream>>>(agg_c, (long)B * Nc * Ch);
    gemm_mlp<128, 128, 64, true, false><<<dim3(Ec / 64, B), dim3(32, 4), 0, stream>>>(
        x, (long)Nc * Cin, x, (long)Nc * Cin, e_coarse, 2L * Ec, Ec,
        wf1e, b1e, agg_c, (long)Nc * Ch);
    gemm_mlp<128, 64, 64, false, false><<<dim3(Nc / 64, B), dim3(32, 4), 0, stream>>>(
        x, (long)Nc * Cin, agg_c, (long)Nc * Ch, nullptr, 0, 0,
        wf1n, b1n, x1, (long)Nc * Ch);

    // 3) knn (coarse -> fine) then both up-pools
    knn_kernel<<<dim3(Nf / 256, B), 256, 0, stream>>>(pos, mask, kidx, kw);
    interp_kernel<128><<<gz((long)B * Nf * 128), 256, 0, stream>>>(x,  kidx, kw, x_up);
    interp_kernel<64> <<<gz((long)B * Nf * 64),  256, 0, stream>>>(x1, kidx, kw, x1_up);

    // 4) skip branch on fine graph -> writes y
    zero_f32<<<gz((long)B * Nf * Cout), 256, 0, stream>>>(agg, (long)B * Nf * Cout);
    gemm_mlp<128, 128, 128, true, false><<<dim3(Ef / 64, B), dim3(32, 8), 0, stream>>>(
        x_up, (long)Nf * Cin, x_up, (long)Nf * Cin, e_fine, 2L * Ef, Ef,
        wfse, bse, agg, (long)Nf * Cout);
    gemm_mlp<128, 128, 128, false, false><<<dim3(Nf / 64, B), dim3(32, 8), 0, stream>>>(
        x_up, (long)Nf * Cin, agg, (long)Nf * Cout, nullptr, 0, 0,
        wfsn, bsn, yout, (long)Nf * Cout);

    // 5) main branch on fine graph -> accumulates into y
    zero_f32<<<gz((long)B * Nf * Cout), 256, 0, stream>>>(agg, (long)B * Nf * Cout);
    gemm_mlp<64, 64, 128, true, false><<<dim3(Ef / 64, B), dim3(32, 8), 0, stream>>>(
        x1_up, (long)Nf * Ch, x1_up, (long)Nf * Ch, e_fine, 2L * Ef, Ef,
        wf2e, b2e, agg, (long)Nf * Cout);
    gemm_mlp<64, 128, 128, false, true><<<dim3(Nf / 64, B), dim3(32, 8), 0, stream>>>(
        x1_up, (long)Nf * Ch, agg, (long)Nf * Cout, nullptr, 0, 0,
        wf2n, b2n, yout, (long)Nf * Cout);

    // 6) batchnorm (training-mode, over all B*Nf rows) + leaky relu, in-place
    zero_f32<<<gz(2L * Cout), 256, 0, stream>>>(sums, 2L * Cout);
    bn_sum_kernel<<<dim3((unsigned)(((long)B * Nf) / 256)), Cout, 0, stream>>>(yout, sums, sumsq);
    bn_final_kernel<<<gz((long)B * Nf * Cout), 256, 0, stream>>>(yout, sums, sumsq, gamma, beta);

    // 7) weight unpool: zeros scatter-set at mask rows
    zero_f32<<<gz((long)B * Nf), 256, 0, stream>>>(wout, (long)B * Nf);
    wfine_kernel<<<gz((long)B * Nc), 256, 0, stream>>>(weights, mask, wout);
}
